// SphericalCTUNOBlock2D_38405597561173
// MI455X (gfx1250) — matrix operand order repaired and checked by reading, not compiled, not run on hardware
//
#include <hip/hip_runtime.h>
#include <hip/hip_bf16.h>
#include <math.h>

// ---------------------------------------------------------------------------
// MI455X / gfx1250 Spherical-FNO block.
// Roofline: wr/wi = 266 MB read-once => ~11.4 us @ 23.3 TB/s dominates; total
// FLOPs ~1.9 GF is negligible. Design: stream w via global_load_async_to_lds
// (512B/instr, ASYNCcnt, split-wait to overlap Wi transfer with Wr compute),
// run every dense contraction on v_wmma_f32_16x16x32_bf16, and keep all
// WMMA-consumed intermediates in B-fragment-packed layout so consumers load
// one v16bf (2x b128) per tile. All short loops force-unrolled so fragment /
// accumulator arrays stay in VGPRs (dynamic indexing would spill to scratch).
// ---------------------------------------------------------------------------

typedef __attribute__((ext_vector_type(16))) __bf16 v16bf;
typedef __attribute__((ext_vector_type(8)))  float  v8f;

#define PI_F 3.14159265358979323846f

__device__ __forceinline__ v8f wmma_bf16(v16bf a, v16bf b, v8f c) {
  return __builtin_amdgcn_wmma_f32_16x16x32_bf16(false, a, false, b,
                                                 (short)0, c, false, false);
}

// --- WMMA register layouts (wave32); A/B k order here is contiguous: k = 16*(lane/16) + e ---
__device__ __forceinline__ int a_row(int lane) { return lane & 15; }
__device__ __forceinline__ int a_kk(int lane, int e) {
  return ((lane >= 16) ? 16 : 0) + e;
}
__device__ __forceinline__ int b_col(int lane) { return lane & 15; }
__device__ __forceinline__ int b_kk(int lane, int e) { return ((lane >= 16) ? 16 : 0) + e; }
__device__ __forceinline__ int d_row(int lane, int r) { return r + ((lane >= 16) ? 8 : 0); }
__device__ __forceinline__ int d_col(int lane) { return lane & 15; }

// --- B-fragment-packed storage: [batch][kt][nt][lane][e], bf16, 32B/frag ---
__device__ __forceinline__ size_t bfrag_idx(int batch, int nkt, int kt, int nt,
                                            int lane, int e) {
  return ((((size_t)batch * nkt + kt) * 4 + nt) * 32 + lane) * 16 + e;
}
__device__ __forceinline__ size_t bfrag_sidx(int batch, int nkt, int k, int n) {
  int kt = k >> 5;
  int lane = ((k >> 4) & 1) * 16 + (n & 15);
  return bfrag_idx(batch, nkt, kt, n >> 4, lane, k & 15);
}

// ---------------------------------------------------------------------------
// k0: psi = t_emb @ dense_w ; bt = psi[64:128] ; bprime[j][i]=psi[j]*time_w[i][j]
// ---------------------------------------------------------------------------
__global__ void k0_setup(const float* __restrict__ t_emb,
                         const float* __restrict__ dense_w,
                         const float* __restrict__ time_w,
                         float* __restrict__ psi, float* __restrict__ bt,
                         __bf16* __restrict__ bprime) {
  int j = threadIdx.x;  // 0..127
  float s = 0.f;
  for (int d = 0; d < 256; ++d) s += t_emb[d] * dense_w[d * 128 + j];
  psi[j] = s;
  __shared__ float sp[128];
  sp[j] = s;
  __syncthreads();
  if (j < 64) bt[j] = sp[64 + j];
  for (int idx = threadIdx.x; idx < 64 * 64; idx += blockDim.x) {
    int jj = idx >> 6, i = idx & 63;
    bprime[bfrag_sidx(0, 2, jj, i)] = (__bf16)(sp[jj] * time_w[i * 64 + jj]);
  }
}

// ---------------------------------------------------------------------------
// k1b: pointwise path. Per theta-row t: H1 = X[t](128x64) @ conv_w + conv_b,
//      H2 = H1 @ bprime + bt  ->  h[t] (f32). 8 waves, wave = 16-row phi tile.
// ---------------------------------------------------------------------------
__global__ void k1b_pointwise(const float* __restrict__ x,
                              const float* __restrict__ conv_w,
                              const float* __restrict__ conv_b,
                              const __bf16* __restrict__ bprime,
                              const float* __restrict__ bt,
                              float* __restrict__ h) {
  int t = blockIdx.x;
  int wave = threadIdx.x >> 5, lane = threadIdx.x & 31;
  int k0 = wave * 16;
  __shared__ __bf16 h1s[8][16 * 64];

  int row = a_row(lane);
  v8f acc[4] = {};
#pragma unroll
  for (int kt = 0; kt < 2; ++kt) {
    v16bf a;
#pragma unroll
    for (int e = 0; e < 16; ++e) {
      int kk = kt * 32 + a_kk(lane, e);
      a[e] = (__bf16)x[((t * 128) + (k0 + row)) * 64 + kk];
    }
#pragma unroll
    for (int nt = 0; nt < 4; ++nt) {
      v16bf b;
#pragma unroll
      for (int e = 0; e < 16; ++e) {
        int kk = kt * 32 + b_kk(lane, e);
        b[e] = (__bf16)conv_w[kk * 64 + nt * 16 + b_col(lane)];
      }
      acc[nt] = wmma_bf16(a, b, acc[nt]);
    }
  }
#pragma unroll
  for (int nt = 0; nt < 4; ++nt) {
    int o = nt * 16 + d_col(lane);
#pragma unroll
    for (int r = 0; r < 8; ++r)
      h1s[wave][d_row(lane, r) * 64 + o] = (__bf16)(acc[nt][r] + conv_b[o]);
  }
  __syncthreads();

  v8f acc2[4] = {};
#pragma unroll
  for (int kt = 0; kt < 2; ++kt) {
    v16bf a;
#pragma unroll
    for (int e = 0; e < 16; ++e) {
      int kk = kt * 32 + a_kk(lane, e);
      a[e] = h1s[wave][row * 64 + kk];
    }
#pragma unroll
    for (int nt = 0; nt < 4; ++nt) {
      v16bf b = *(const v16bf*)&bprime[bfrag_idx(0, 2, kt, nt, lane, 0)];
      acc2[nt] = wmma_bf16(a, b, acc2[nt]);
    }
  }
#pragma unroll
  for (int nt = 0; nt < 4; ++nt) {
    int o = nt * 16 + d_col(lane);
#pragma unroll
    for (int r = 0; r < 8; ++r)
      h[((t * 128) + (k0 + d_row(lane, r))) * 64 + o] = acc2[nt][r] + bt[o];
  }
}

// ---------------------------------------------------------------------------
// k1: forward DFT over longitude. fm[m,t,c] = s_t * sum_k x[t,k,c] e^{-2pi i mi k/128}
//     Output fragment-packed for k2 (batch=m, K=t (nkt=4), N=c).
// ---------------------------------------------------------------------------
__global__ void k1_dft(const float* __restrict__ x,
                       const float* __restrict__ quad_w,
                       __bf16* __restrict__ fmr, __bf16* __restrict__ fmi) {
  int t = blockIdx.x;
  int wave = threadIdx.x >> 5, lane = threadIdx.x & 31;
  __shared__ __bf16 xb[128 * 64];
  for (int i = threadIdx.x; i < 128 * 64; i += blockDim.x)
    xb[i] = (__bf16)x[t * (128 * 64) + i];
  __syncthreads();

  int m0 = wave * 16;
  float s = (2.0f * PI_F / 128.0f) * quad_w[t];
  int row = a_row(lane);
  float mi = (float)(m0 + row - 63);
  v8f accr[4] = {}, acci[4] = {};
#pragma unroll
  for (int kt = 0; kt < 4; ++kt) {
    v16bf ac, as;
#pragma unroll
    for (int e = 0; e < 16; ++e) {
      int kk = kt * 32 + a_kk(lane, e);
      float ang = -2.0f * PI_F * mi * (float)kk * (1.0f / 128.0f);
      float sv, cv;
      __sincosf(ang, &sv, &cv);
      ac[e] = (__bf16)cv;
      as[e] = (__bf16)sv;  // e^{i*ang} = cos + i sin, ang already has the minus
    }
#pragma unroll
    for (int nt = 0; nt < 4; ++nt) {
      v16bf b;
#pragma unroll
      for (int e = 0; e < 16; ++e) {
        int kk = kt * 32 + b_kk(lane, e);
        b[e] = xb[kk * 64 + nt * 16 + b_col(lane)];
      }
      accr[nt] = wmma_bf16(ac, b, accr[nt]);
      acci[nt] = wmma_bf16(as, b, acci[nt]);
    }
  }
#pragma unroll
  for (int nt = 0; nt < 4; ++nt) {
    int c = nt * 16 + d_col(lane);
#pragma unroll
    for (int r = 0; r < 8; ++r) {
      int mm = m0 + d_row(lane, r);
      if (mm < 127) {
        fmr[bfrag_sidx(mm, 4, t, c)] = (__bf16)(s * accr[nt][r]);
        fmi[bfrag_sidx(mm, 4, t, c)] = (__bf16)(s * acci[nt][r]);
      }
    }
  }
}

// ---------------------------------------------------------------------------
// k2: forward Legendre. Per m: Flm[l,c] = sum_t P_in[l,m,t] fm[t,c]. 4 waves.
// ---------------------------------------------------------------------------
__global__ void k2_leg_in(const __bf16* __restrict__ fmr,
                          const __bf16* __restrict__ fmi,
                          const float* __restrict__ P_in,
                          __bf16* __restrict__ flmr, __bf16* __restrict__ flmi) {
  int m = blockIdx.x;
  int wave = threadIdx.x >> 5, lane = threadIdx.x & 31;
  int l0 = wave * 16;
  int row = a_row(lane);
  v8f ar[4] = {}, ai[4] = {};
#pragma unroll
  for (int kt = 0; kt < 4; ++kt) {
    v16bf a;
#pragma unroll
    for (int e = 0; e < 16; ++e) {
      int tt = kt * 32 + a_kk(lane, e);
      a[e] = (__bf16)P_in[((l0 + row) * 127 + m) * 128 + tt];
    }
#pragma unroll
    for (int nt = 0; nt < 4; ++nt) {
      v16bf br = *(const v16bf*)&fmr[bfrag_idx(m, 4, kt, nt, lane, 0)];
      v16bf bi = *(const v16bf*)&fmi[bfrag_idx(m, 4, kt, nt, lane, 0)];
      ar[nt] = wmma_bf16(a, br, ar[nt]);
      ai[nt] = wmma_bf16(a, bi, ai[nt]);
    }
  }
#pragma unroll
  for (int nt = 0; nt < 4; ++nt) {
    int c = nt * 16 + d_col(lane);
#pragma unroll
    for (int r = 0; r < 8; ++r) {
      int l = l0 + d_row(lane, r);
      flmr[(m * 64 + l) * 64 + c] = (__bf16)ar[nt][r];
      flmi[(m * 64 + l) * 64 + c] = (__bf16)ai[nt][r];
    }
  }
}

// ---------------------------------------------------------------------------
// k3: spectral weights (the 266MB roofline stage). One wave per (l,m).
//     wr/wi streamed HBM->LDS with global_load_async_to_lds_b128 (ASYNCcnt).
//     Issue order: 32x Wr then 32x Wi; async loads complete in order, so
//     s_wait_asynccnt 32 releases the Wr pass while Wi is still in flight.
//     A fragments kept in named registers (aA0/aA1) under unrolled loops.
// ---------------------------------------------------------------------------
#define K3_WAVES 2
__global__ void k3_weights(const __bf16* __restrict__ flmr,
                           const __bf16* __restrict__ flmi,
                           const float* __restrict__ wr,
                           const float* __restrict__ wi,
                           __bf16* __restrict__ hlmr, __bf16* __restrict__ hlmi) {
  int m = blockIdx.x;
  int wave = threadIdx.x >> 5, lane = threadIdx.x & 31;
  int l = blockIdx.y * K3_WAVES + wave;
  __shared__ float wbuf[K3_WAVES][2][64 * 64];  // 64 KB: f32 Wr|Wi per wave

  // Async-stream this (l,m)'s Wr then Wi (16KB each) into LDS.
  unsigned gbase = (unsigned)((l * 127 + m) * 16384u);  // byte offset
  unsigned ldsr0 = (unsigned)(size_t)&wbuf[wave][0][0];
  unsigned ldsi0 = (unsigned)(size_t)&wbuf[wave][1][0];
#pragma unroll 4
  for (int j = 0; j < 32; ++j) {
    unsigned off16 = (unsigned)(j * 32 + lane) * 16u;
    asm volatile("global_load_async_to_lds_b128 %0, %1, %2"
                 :: "v"(ldsr0 + off16), "v"(gbase + off16), "s"(wr) : "memory");
  }
#pragma unroll 4
  for (int j = 0; j < 32; ++j) {
    unsigned off16 = (unsigned)(j * 32 + lane) * 16u;
    asm volatile("global_load_async_to_lds_b128 %0, %1, %2"
                 :: "v"(ldsi0 + off16), "v"(gbase + off16), "s"(wi) : "memory");
  }

  // Build both A fragments once into named registers (reused for both passes).
  int row = a_row(lane);
  v16bf aA0, aA1;
#pragma unroll
  for (int e = 0; e < 16; ++e) {
    int i0 = a_kk(lane, e);
    __bf16 v0 = (__bf16)0.0f, v1 = (__bf16)0.0f;
    if (row == 0) {
      v0 = flmr[(m * 64 + l) * 64 + i0];
      v1 = flmr[(m * 64 + l) * 64 + 32 + i0];
    } else if (row == 1) {
      v0 = flmi[(m * 64 + l) * 64 + i0];
      v1 = flmi[(m * 64 + l) * 64 + 32 + i0];
    }
    aA0[e] = v0;
    aA1[e] = v1;
  }

  const float* wrL = &wbuf[wave][0][0];
  const float* wiL = &wbuf[wave][1][0];
  v8f aWr[4] = {}, aWi[4] = {};

  // Wr pass: only the first 32 async ops must have landed.
  asm volatile("s_wait_asynccnt 0x20" ::: "memory");
#pragma unroll
  for (int kt = 0; kt < 2; ++kt) {
    v16bf a = (kt == 0) ? aA0 : aA1;
#pragma unroll
    for (int nt = 0; nt < 4; ++nt) {
      v16bf br;
#pragma unroll
      for (int e = 0; e < 16; ++e) {
        int i = kt * 32 + b_kk(lane, e);
        br[e] = (__bf16)wrL[i * 64 + nt * 16 + b_col(lane)];
      }
      aWr[nt] = wmma_bf16(a, br, aWr[nt]);
    }
  }
  // Wi pass: now require everything.
  asm volatile("s_wait_asynccnt 0x0" ::: "memory");
#pragma unroll
  for (int kt = 0; kt < 2; ++kt) {
    v16bf a = (kt == 0) ? aA0 : aA1;
#pragma unroll
    for (int nt = 0; nt < 4; ++nt) {
      v16bf bi;
#pragma unroll
      for (int e = 0; e < 16; ++e) {
        int i = kt * 32 + b_kk(lane, e);
        bi[e] = (__bf16)wiL[i * 64 + nt * 16 + b_col(lane)];
      }
      aWi[nt] = wmma_bf16(a, bi, aWi[nt]);
    }
  }

  if (lane < 16) {
#pragma unroll
    for (int nt = 0; nt < 4; ++nt) {
      int o = nt * 16 + lane;
      float fr_wr = aWr[nt][0], fi_wr = aWr[nt][1];
      float fr_wi = aWi[nt][0], fi_wi = aWi[nt][1];
      // fragment-packed for k4 (batch=m, K=l (nkt=2), N=o)
      hlmr[bfrag_sidx(m, 2, l, o)] = (__bf16)(fr_wr - fi_wi);
      hlmi[bfrag_sidx(m, 2, l, o)] = (__bf16)(fr_wi + fi_wr);
    }
  }
}

// ---------------------------------------------------------------------------
// k4: inverse Legendre. Per m: g[t,o] = sum_l P_out[l,m,t] hlm[l,o]. 8 waves.
//     g written fragment-packed for k5 (batch=t, K=m padded to 128, N=o).
// ---------------------------------------------------------------------------
__global__ void k4_leg_out(const __bf16* __restrict__ hlmr,
                           const __bf16* __restrict__ hlmi,
                           const float* __restrict__ P_out,
                           __bf16* __restrict__ gr, __bf16* __restrict__ gi) {
  int m = blockIdx.x;
  int wave = threadIdx.x >> 5, lane = threadIdx.x & 31;
  int t0 = wave * 16;
  int row = a_row(lane);
  v8f ar[4] = {}, ai[4] = {};
#pragma unroll
  for (int kt = 0; kt < 2; ++kt) {
    v16bf a;
#pragma unroll
    for (int e = 0; e < 16; ++e) {
      int l = kt * 32 + a_kk(lane, e);
      a[e] = (__bf16)P_out[(l * 127 + m) * 128 + (t0 + row)];
    }
#pragma unroll
    for (int nt = 0; nt < 4; ++nt) {
      v16bf br = *(const v16bf*)&hlmr[bfrag_idx(m, 2, kt, nt, lane, 0)];
      v16bf bi = *(const v16bf*)&hlmi[bfrag_idx(m, 2, kt, nt, lane, 0)];
      ar[nt] = wmma_bf16(a, br, ar[nt]);
      ai[nt] = wmma_bf16(a, bi, ai[nt]);
    }
  }
#pragma unroll
  for (int nt = 0; nt < 4; ++nt) {
    int o = nt * 16 + d_col(lane);
#pragma unroll
    for (int r = 0; r < 8; ++r) {
      int t = t0 + d_row(lane, r);
      gr[bfrag_sidx(t, 4, m, o)] = (__bf16)ar[nt][r];
      gi[bfrag_sidx(t, 4, m, o)] = (__bf16)ai[nt][r];
    }
  }
}

// ---------------------------------------------------------------------------
// k5: inverse DFT over m + add pointwise path + LayerNorm + ReLU (fused).
// ---------------------------------------------------------------------------
__global__ void k5_final(const __bf16* __restrict__ gr,
                         const __bf16* __restrict__ gi,
                         const float* __restrict__ h,
                         const float* __restrict__ ln_scale,
                         const float* __restrict__ ln_bias,
                         float* __restrict__ out) {
  int t = blockIdx.x;
  int wave = threadIdx.x >> 5, lane = threadIdx.x & 31;
  int k0 = wave * 16;
  __shared__ float tile[8][16 * 64];
  int row = a_row(lane);
  float phi = 2.0f * PI_F * (float)(k0 + row) * (1.0f / 128.0f);
  v8f acc[4] = {};
#pragma unroll
  for (int kt = 0; kt < 4; ++kt) {
    v16bf ac, as;
#pragma unroll
    for (int e = 0; e < 16; ++e) {
      int mm = kt * 32 + a_kk(lane, e);
      float mi = (float)(mm - 63);
      float sv, cv;
      __sincosf(mi * phi, &sv, &cv);
      if (mm >= 127) { cv = 0.f; sv = 0.f; }  // pad row
      ac[e] = (__bf16)cv;
      as[e] = (__bf16)(-sv);  // Re(g e^{i mi phi}) = gr*cos - gi*sin
    }
#pragma unroll
    for (int nt = 0; nt < 4; ++nt) {
      v16bf br = *(const v16bf*)&gr[bfrag_idx(t, 4, kt, nt, lane, 0)];
      v16bf bi = *(const v16bf*)&gi[bfrag_idx(t, 4, kt, nt, lane, 0)];
      if (kt == 3 && lane >= 16) {  // K element m=127 never written: zero it
        br[15] = (__bf16)0.0f;
        bi[15] = (__bf16)0.0f;
      }
      acc[nt] = wmma_bf16(ac, br, acc[nt]);
      acc[nt] = wmma_bf16(as, bi, acc[nt]);
    }
  }
#pragma unroll
  for (int nt = 0; nt < 4; ++nt) {
    int o = nt * 16 + d_col(lane);
#pragma unroll
    for (int r = 0; r < 8; ++r) {
      int rw = d_row(lane, r);
      tile[wave][rw * 64 + o] = acc[nt][r] + h[((t * 128) + k0 + rw) * 64 + o];
    }
  }
  __syncthreads();
  if (lane < 16) {
    const float* v = &tile[wave][lane * 64];
    float mu = 0.f;
    for (int o = 0; o < 64; ++o) mu += v[o];
    mu *= (1.0f / 64.0f);
    float var = 0.f;
    for (int o = 0; o < 64; ++o) { float d = v[o] - mu; var += d * d; }
    var *= (1.0f / 64.0f);
    float inv = rsqrtf(var + 1e-6f);
    for (int o = 0; o < 64; ++o) {
      float y = (v[o] - mu) * inv * ln_scale[o] + ln_bias[o];
      out[((t * 128) + k0 + lane) * 64 + o] = fmaxf(y, 0.f);
    }
  }
}

// ---------------------------------------------------------------------------
extern "C" void kernel_launch(void* const* d_in, const int* in_sizes, int n_in,
                              void* d_out, int out_size, void* d_ws, size_t ws_size,
                              hipStream_t stream) {
  const float* x       = (const float*)d_in[0];
  const float* t_emb   = (const float*)d_in[1];
  const float* wr      = (const float*)d_in[2];
  const float* wi      = (const float*)d_in[3];
  const float* conv_w  = (const float*)d_in[4];
  const float* conv_b  = (const float*)d_in[5];
  const float* time_w  = (const float*)d_in[6];
  const float* dense_w = (const float*)d_in[7];
  const float* ln_s    = (const float*)d_in[8];
  const float* ln_b    = (const float*)d_in[9];
  const float* P_in    = (const float*)d_in[10];
  const float* P_out   = (const float*)d_in[11];
  const float* quad_w  = (const float*)d_in[12];

  char* ws = (char*)d_ws;
  float*  psi    = (float*)(ws + 0);                 // 128 f32
  float*  bt     = (float*)(ws + 512);               // 64 f32
  __bf16* bprime = (__bf16*)(ws + 1024);             // 4096 bf16 (frag)
  __bf16* fmr  = (__bf16*)(ws + (1 << 14));          // 127*8192 (frag)
  __bf16* fmi  = fmr  + 127 * 8192;
  __bf16* flmr = fmi  + 127 * 8192;                  // 127*64*64 linear
  __bf16* flmi = flmr + 127 * 4096;
  __bf16* hlmr = flmi + 127 * 4096;                  // 127*4096 (frag, nkt=2)
  __bf16* hlmi = hlmr + 127 * 4096;
  __bf16* grr  = hlmi + 127 * 4096;                  // 128*8192 (frag, padded K)
  __bf16* gii  = grr  + 128 * 8192;
  float*  h    = (float*)(gii + 128 * 8192);         // [128][128][64] f32

  k0_setup<<<1, 128, 0, stream>>>(t_emb, dense_w, time_w, psi, bt, bprime);
  k1b_pointwise<<<128, 256, 0, stream>>>(x, conv_w, conv_b, bprime, bt, h);
  k1_dft<<<128, 256, 0, stream>>>(x, quad_w, fmr, fmi);
  k2_leg_in<<<127, 128, 0, stream>>>(fmr, fmi, P_in, flmr, flmi);
  k3_weights<<<dim3(127, 64 / K3_WAVES), K3_WAVES * 32, 0, stream>>>(
      flmr, flmi, wr, wi, hlmr, hlmi);
  k4_leg_out<<<127, 256, 0, stream>>>(hlmr, hlmi, P_out, grr, gii);
  k5_final<<<128, 256, 0, stream>>>(grr, gii, h, ln_s, ln_b, (float*)d_out);
}
